// MoELayer_10514079940880
// MI455X (gfx1250) — matrix operand order repaired and checked by reading, not compiled, not run on hardware
//
#include <hip/hip_runtime.h>

// Problem constants (match reference)
constexpr int BT  = 2048;            // B*T tokens
constexpr int DD  = 2048;            // model dim
constexpr int FF  = 768;             // ffn dim
constexpr int EE  = 16;              // experts
constexpr int CAP = BT * 2;          // total (token,k) slots = 4096

// GEMM tiling
constexpr int TBM = 64;              // token rows per block
constexpr int TBN = 128;             // out cols per block
constexpr int TBK = 32;              // K chunk (WMMA K)
constexpr int LDP = TBK + 8;         // padded LDS row (80B stride: 16B aligned, conflict-free)
constexpr int MT1 = CAP / TBM;       // 64 worst-case token tiles per expert
constexpr int NT1 = FF / TBN;        // 6
constexpr int NT2 = DD / TBN;        // 16

typedef __attribute__((ext_vector_type(16))) __bf16         v16bf;
typedef __attribute__((ext_vector_type(8)))  float          v8f;
typedef __attribute__((ext_vector_type(8)))  unsigned short ushort8;

union FragAB { v16bf bf; ushort8 us[2]; };

// Native hardware conversion (lowers to v_cvt_*bf16_f32 on gfx1250)
__device__ __forceinline__ unsigned short f32_to_bf16(float f) {
  __bf16 h = (__bf16)f;
  return __builtin_bit_cast(unsigned short, h);
}
__device__ __forceinline__ unsigned pack_bf16x2(float lo, float hi) {
  return (unsigned)f32_to_bf16(lo) | ((unsigned)f32_to_bf16(hi) << 16);
}

// ---------------------------------------------------------------- kernel 0
__global__ void moe_zero(int* __restrict__ cnt) {
  if (threadIdx.x < EE) cnt[threadIdx.x] = 0;
}

// ---------------------------------------------------------------- kernel 1: router + top-2 + list build
__global__ __launch_bounds__(256) void moe_router(
    const float* __restrict__ x, const float* __restrict__ rw,
    int* __restrict__ cnt, int* __restrict__ list, float* __restrict__ wbuf)
{
  const int lane = threadIdx.x & 31;
  const int wave = threadIdx.x >> 5;
  const int t = blockIdx.x * 8 + wave;
  if (t >= BT) return;

  float acc[EE];
  #pragma unroll
  for (int e = 0; e < EE; ++e) acc[e] = 0.f;

  const float* xr = x + (size_t)t * DD;
  for (int d = lane; d < DD; d += 32) {
    const float xv = xr[d];
    const float* r = rw + (size_t)d * EE;
    #pragma unroll
    for (int e = 0; e < EE; ++e) acc[e] = fmaf(xv, r[e], acc[e]);
  }
  #pragma unroll
  for (int e = 0; e < EE; ++e) {
    #pragma unroll
    for (int off = 16; off > 0; off >>= 1) acc[e] += __shfl_xor(acc[e], off, 32);
  }
  if (lane == 0) {
    int i0 = 0; float s0 = acc[0];
    #pragma unroll
    for (int e = 1; e < EE; ++e) if (acc[e] > s0) { s0 = acc[e]; i0 = e; }
    int i1 = (i0 == 0) ? 1 : 0; float s1 = acc[i1];
    #pragma unroll
    for (int e = 0; e < EE; ++e)
      if (e != i0 && acc[e] > s1) { s1 = acc[e]; i1 = e; }
    // softmax top-2 renormalized: w0 = e^s0/(e^s0+e^s1)
    const float w0 = 1.f / (1.f + __expf(s1 - s0));
    const float w1 = 1.f - w0;
    int sl0 = atomicAdd(&cnt[i0], 1);
    list[i0 * CAP + sl0] = t * 2 + 0;
    wbuf[t * 2 + 0] = w0;
    int sl1 = atomicAdd(&cnt[i1], 1);
    list[i1 * CAP + sl1] = t * 2 + 1;
    wbuf[t * 2 + 1] = w1;
  }
}

// ---------------------------------------------------------------- kernel 2: gate/up GEMM + SwiGLU
// Double-buffered LDS pipeline: load tile k+1 while WMMAs consume tile k.
__global__ __launch_bounds__(256) void moe_gateup(
    const float* __restrict__ x, const float* __restrict__ gw,
    const float* __restrict__ uw, const int* __restrict__ cnt,
    const int* __restrict__ list, unsigned short* __restrict__ Hbuf)
{
  const int bx = blockIdx.x;
  const int e  = bx / (MT1 * NT1);
  const int r0 = bx % (MT1 * NT1);
  const int mt = r0 / NT1;
  const int nt = r0 % NT1;
  const int ce = cnt[e];
  const int m0 = mt * TBM;
  if (m0 >= ce) return;

  __shared__ alignas(16) unsigned short sA [2][TBM][LDP];
  __shared__ alignas(16) unsigned short sBg[2][TBN][LDP];
  __shared__ alignas(16) unsigned short sBu[2][TBN][LDP];
  __shared__ int sTk[TBM];

  const int tid = threadIdx.x;
  if (tid < TBM) {
    const int j = m0 + tid;
    sTk[tid] = (j < ce) ? list[e * CAP + j] : -1;
  }
  __syncthreads();

  const int lane = tid & 31, wid = tid >> 5;
  const int wm = wid >> 2, wn = wid & 3;   // 2x4 wave grid -> 64x128 block tile

  const v8f vz = {0.f, 0.f, 0.f, 0.f, 0.f, 0.f, 0.f, 0.f};
  v8f accG[2][2], accU[2][2];
  #pragma unroll
  for (int i = 0; i < 2; ++i)
    #pragma unroll
    for (int j = 0; j < 2; ++j) { accG[i][j] = vz; accU[i][j] = vz; }

  // A loader: one row, 8 consecutive cols. B loader: K-pair, 8 consecutive cols
  const int lrow = tid >> 2, lcg = (tid & 3) * 8;
  const int lk2  = (tid & 15) * 2, lnb = (tid >> 4) * 8;
  const float* gbase = gw + (size_t)e * DD * FF + (size_t)nt * TBN + lnb;
  const float* ubase = uw + (size_t)e * DD * FF + (size_t)nt * TBN + lnb;
  const int koffA = (lane & 16) ? 16 : 0;
  const int koffB = (lane & 16) ? 16 : 0;
  const int tkL = sTk[lrow];
  const float* asrc = (tkL >= 0) ? x + (size_t)(tkL >> 1) * DD + lcg : nullptr;

  auto load_tiles = [&](int k0, int buf) {
    // A tile: gather token rows, cvt f32->bf16, one 16B LDS store
    ushort8 pa;
    if (tkL >= 0) {
      const float* src = asrc + k0;
      #pragma unroll
      for (int i = 0; i < 8; ++i) pa[i] = f32_to_bf16(src[i]);
    } else {
      #pragma unroll
      for (int i = 0; i < 8; ++i) pa[i] = 0;
    }
    *(ushort8*)&sA[buf][lrow][lcg] = pa;
    // B tiles transposed into LDS (sB[n][k]); packed bf16x2 dword stores
    const float* g0 = gbase + (size_t)(k0 + lk2) * FF;
    const float* g1 = g0 + FF;
    const float* u0 = ubase + (size_t)(k0 + lk2) * FF;
    const float* u1 = u0 + FF;
    #pragma unroll
    for (int i = 0; i < 8; ++i) {
      *(unsigned*)&sBg[buf][lnb + i][lk2] = pack_bf16x2(g0[i], g1[i]);
      *(unsigned*)&sBu[buf][lnb + i][lk2] = pack_bf16x2(u0[i], u1[i]);
    }
  };

  load_tiles(0, 0);
  __syncthreads();

  constexpr int KITERS = DD / TBK;
  for (int it = 0; it < KITERS; ++it) {
    const int buf = it & 1;
    if (it + 1 < KITERS) load_tiles((it + 1) * TBK, buf ^ 1);

    FragAB a[2], bg[2], bu[2];
    #pragma unroll
    for (int fm = 0; fm < 2; ++fm) {
      const int r = wm * 32 + fm * 16 + (lane & 15);
      a[fm].us[0] = *(const ushort8*)&sA[buf][r][koffA];
      a[fm].us[1] = *(const ushort8*)&sA[buf][r][koffA + 8];
    }
    #pragma unroll
    for (int fn = 0; fn < 2; ++fn) {
      const int c = wn * 32 + fn * 16 + (lane & 15);
      bg[fn].us[0] = *(const ushort8*)&sBg[buf][c][koffB];
      bg[fn].us[1] = *(const ushort8*)&sBg[buf][c][koffB + 8];
      bu[fn].us[0] = *(const ushort8*)&sBu[buf][c][koffB];
      bu[fn].us[1] = *(const ushort8*)&sBu[buf][c][koffB + 8];
    }
    #pragma unroll
    for (int fm = 0; fm < 2; ++fm)
      #pragma unroll
      for (int fn = 0; fn < 2; ++fn) {
        accG[fm][fn] = __builtin_amdgcn_wmma_f32_16x16x32_bf16(
            false, a[fm].bf, false, bg[fn].bf, (short)0, accG[fm][fn], false, false);
        accU[fm][fn] = __builtin_amdgcn_wmma_f32_16x16x32_bf16(
            false, a[fm].bf, false, bu[fn].bf, (short)0, accU[fm][fn], false, false);
      }
    __syncthreads();
  }

  // epilogue: h = silu(g)*u -> bf16 Hbuf[(t*2+k)*F + n]
  const int mhalf = (lane >> 4) << 3;
  const int ncl = lane & 15;
  #pragma unroll
  for (int fm = 0; fm < 2; ++fm)
    #pragma unroll
    for (int fn = 0; fn < 2; ++fn)
      #pragma unroll
      for (int i = 0; i < 8; ++i) {
        const int m = wm * 32 + fm * 16 + mhalf + i;
        const int tk = sTk[m];
        if (tk >= 0) {
          const float gv = accG[fm][fn][i];
          const float uv = accU[fm][fn][i];
          const float h = (gv / (1.f + __expf(-gv))) * uv;
          const int n = nt * TBN + wn * 32 + fn * 16 + ncl;
          Hbuf[(size_t)tk * FF + n] = f32_to_bf16(h);
        }
      }
}

// ---------------------------------------------------------------- kernel 3: down GEMM (weighted)
__global__ __launch_bounds__(256) void moe_down(
    const unsigned short* __restrict__ Hbuf, const float* __restrict__ dw,
    const int* __restrict__ cnt, const int* __restrict__ list,
    const float* __restrict__ wbuf, float* __restrict__ Ybuf)
{
  const int bx = blockIdx.x;
  const int e  = bx / (MT1 * NT2);
  const int r0 = bx % (MT1 * NT2);
  const int mt = r0 / NT2;
  const int nt = r0 % NT2;
  const int ce = cnt[e];
  const int m0 = mt * TBM;
  if (m0 >= ce) return;

  __shared__ alignas(16) unsigned short sA[2][TBM][LDP];
  __shared__ alignas(16) unsigned short sB[2][TBN][LDP];
  __shared__ int   sTk[TBM];
  __shared__ float sW [TBM];

  const int tid = threadIdx.x;
  if (tid < TBM) {
    const int j = m0 + tid;
    const int tk = (j < ce) ? list[e * CAP + j] : -1;
    sTk[tid] = tk;
    sW[tid]  = (tk >= 0) ? wbuf[tk] : 0.f;
  }
  __syncthreads();

  const int lane = tid & 31, wid = tid >> 5;
  const int wm = wid >> 2, wn = wid & 3;

  const v8f vz = {0.f, 0.f, 0.f, 0.f, 0.f, 0.f, 0.f, 0.f};
  v8f acc[2][2];
  #pragma unroll
  for (int i = 0; i < 2; ++i)
    #pragma unroll
    for (int j = 0; j < 2; ++j) acc[i][j] = vz;

  const int lrow = tid >> 2, lcg = (tid & 3) * 8;
  const int lk2  = (tid & 15) * 2, lnb = (tid >> 4) * 8;
  const float* dbase = dw + (size_t)e * FF * DD + (size_t)nt * TBN + lnb;
  const int koffA = (lane & 16) ? 16 : 0;
  const int koffB = (lane & 16) ? 16 : 0;
  const int tkL = sTk[lrow];
  const unsigned short* asrc = (tkL >= 0) ? Hbuf + (size_t)tkL * FF + lcg : nullptr;

  auto load_tiles = [&](int k0, int buf) {
    if (tkL >= 0) {
      *(ushort8*)&sA[buf][lrow][lcg] = *(const ushort8*)(asrc + k0);
    } else {
      const ushort8 z = {0, 0, 0, 0, 0, 0, 0, 0};
      *(ushort8*)&sA[buf][lrow][lcg] = z;
    }
    const float* d0 = dbase + (size_t)(k0 + lk2) * DD;
    const float* d1 = d0 + DD;
    #pragma unroll
    for (int i = 0; i < 8; ++i)
      *(unsigned*)&sB[buf][lnb + i][lk2] = pack_bf16x2(d0[i], d1[i]);
  };

  load_tiles(0, 0);
  __syncthreads();

  constexpr int KITERS = FF / TBK;
  for (int it = 0; it < KITERS; ++it) {
    const int buf = it & 1;
    if (it + 1 < KITERS) load_tiles((it + 1) * TBK, buf ^ 1);

    FragAB a[2], b[2];
    #pragma unroll
    for (int fm = 0; fm < 2; ++fm) {
      const int r = wm * 32 + fm * 16 + (lane & 15);
      a[fm].us[0] = *(const ushort8*)&sA[buf][r][koffA];
      a[fm].us[1] = *(const ushort8*)&sA[buf][r][koffA + 8];
    }
    #pragma unroll
    for (int fn = 0; fn < 2; ++fn) {
      const int c = wn * 32 + fn * 16 + (lane & 15);
      b[fn].us[0] = *(const ushort8*)&sB[buf][c][koffB];
      b[fn].us[1] = *(const ushort8*)&sB[buf][c][koffB + 8];
    }
    #pragma unroll
    for (int fm = 0; fm < 2; ++fm)
      #pragma unroll
      for (int fn = 0; fn < 2; ++fn)
        acc[fm][fn] = __builtin_amdgcn_wmma_f32_16x16x32_bf16(
            false, a[fm].bf, false, b[fn].bf, (short)0, acc[fm][fn], false, false);
    __syncthreads();
  }

  const int mhalf = (lane >> 4) << 3;
  const int ncl = lane & 15;
  #pragma unroll
  for (int fm = 0; fm < 2; ++fm)
    #pragma unroll
    for (int fn = 0; fn < 2; ++fn)
      #pragma unroll
      for (int i = 0; i < 8; ++i) {
        const int m = wm * 32 + fm * 16 + mhalf + i;
        const int tk = sTk[m];
        if (tk >= 0) {
          const int n = nt * TBN + wn * 32 + fn * 16 + ncl;
          Ybuf[(size_t)tk * DD + n] = sW[m] * acc[fm][fn][i];
        }
      }
}

// ---------------------------------------------------------------- kernel 4: deterministic combine
__global__ __launch_bounds__(256) void moe_combine(
    const float* __restrict__ Ybuf, float* __restrict__ out)
{
  const int i = blockIdx.x * 256 + threadIdx.x;     // i = t*D + d, over BT*DD
  const int t = i >> 11;                            // D = 2048
  const size_t base = (size_t)i + ((size_t)t << 11);
  out[i] = Ybuf[base] + Ybuf[base + DD];
}

// ---------------------------------------------------------------- host launcher
extern "C" void kernel_launch(void* const* d_in, const int* in_sizes, int n_in,
                              void* d_out, int out_size, void* d_ws, size_t ws_size,
                              hipStream_t stream)
{
  (void)in_sizes; (void)n_in; (void)out_size; (void)ws_size;
  const float* x  = (const float*)d_in[0];
  const float* rw = (const float*)d_in[1];
  const float* gw = (const float*)d_in[2];
  const float* uw = (const float*)d_in[3];
  const float* dw = (const float*)d_in[4];
  float* out = (float*)d_out;
  char* ws = (char*)d_ws;

  // workspace layout (all 256B aligned): ~40.1 MB total
  int*            cnt  = (int*)(ws + 0);
  int*            list = (int*)(ws + 256);                       // E*CAP ints  = 256 KB
  float*          wbuf = (float*)(ws + 262400);                  // CAP floats  = 16 KB
  unsigned short* Hbuf = (unsigned short*)(ws + 278784);         // CAP*F bf16  = 6.29 MB
  float*          Ybuf = (float*)(ws + 6570240);                 // CAP*D f32   = 33.55 MB

  moe_zero   <<<1, 32, 0, stream>>>(cnt);
  moe_router <<<BT / 8, 256, 0, stream>>>(x, rw, cnt, list, wbuf);
  moe_gateup <<<EE * MT1 * NT1, 256, 0, stream>>>(x, gw, uw, cnt, list, Hbuf);
  moe_down   <<<EE * MT1 * NT2, 256, 0, stream>>>(Hbuf, dw, cnt, list, wbuf, Ybuf);
  moe_combine<<<(BT * DD) / 256, 256, 0, stream>>>(Ybuf, out);
}
